// MultiHeadAttention_87840671138396
// MI455X (gfx1250) — matrix operand order repaired and checked by reading, not compiled, not run on hardware
//
#include <hip/hip_runtime.h>
#include <cstdint>

// ---------------------------------------------------------------------------
// MI455X (gfx1250) multi-head attention forward with Kerple bias.
// B=8 S=1024 D=512 H=8 DK=64.
//
// Roofline: ~35 GFLOP vs a mandatory 268MB attn f32 write -> HBM-write bound
// (~14us floor @ 23.3 TB/s). Strategy: f16-in/f32-acc WMMA for all GEMMs,
// keep the 32x1024 score panel in LDS (320KB/WGP), skip tiles above the
// causal diagonal, write attn to HBM exactly once (b128 stores), and stage
// V tiles with gfx1250 async global->LDS DMA (ASYNCcnt) pipelined against
// the WMMA stream.
// ---------------------------------------------------------------------------

#define B_DIM 8
#define S_DIM 1024
#define D_DIM 512
#define H_DIM 8
#define DK_DIM 64

typedef __attribute__((ext_vector_type(16))) _Float16 v16h;
typedef __attribute__((ext_vector_type(8)))  _Float16 v8h;
typedef __attribute__((ext_vector_type(4)))  _Float16 v4h;
typedef __attribute__((ext_vector_type(8)))  float    v8f;
typedef __attribute__((ext_vector_type(4)))  float    v4f;

// Shared-aperture flat pointers carry the LDS byte address in the low 32 bits.
__device__ __forceinline__ unsigned lds_off(const void* p) {
  return (unsigned)(uintptr_t)p;
}

// A fragment (16x32, MxK, f16): lane<16 holds row M=lane, K ko..ko+15
// with ko=0; lane>=16 holds same row set with ko=16. Two contiguous 16B loads.
__device__ __forceinline__ v16h load_a_frag(const _Float16* base, int stride, int lane) {
  const int r  = lane & 15;
  const int ko = (lane >> 4) * 16;
  const _Float16* p = base + r * stride + ko;
  union { v16h v; v8h h[2]; } u;
  u.h[0] = *(const v8h*)(p);
  u.h[1] = *(const v8h*)(p + 8);
  return u.v;
}

// B fragment (32x16, KxN, f16) from a [N][K] row-major tile:
// lane<16 -> n=lane, K=0..15 ; lane>=16 -> n=lane-16, K=16..31. One 32B load.
__device__ __forceinline__ v16h load_b_frag(const _Float16* base, int stride, int lane) {
  const int n  = lane & 15;
  const int ko = (lane >> 4) * 16;
  return *(const v16h*)(base + n * stride + ko);
}

// Same, but tolerating a row stride that is only 16B-aligned (padded vT panel).
__device__ __forceinline__ v16h load_b_frag_u(const _Float16* base, int stride, int lane) {
  const int n  = lane & 15;
  const int ko = (lane >> 4) * 16;
  const _Float16* p = base + n * stride + ko;
  union { v16h v; v8h h[2]; } u;
  u.h[0] = *(const v8h*)(p);
  u.h[1] = *(const v8h*)(p + 8);
  return u.v;
}

__device__ __forceinline__ v8f wmma16(v16h a, v16h b, v8f c) {
  return __builtin_amdgcn_wmma_f32_16x16x32_f16(false, a, false, b, (short)0, c,
                                                false, false);
}

// f32 tile (rows x 32) -> f16 LDS, vectorized: v4f global load, 8B LDS store.
__device__ __forceinline__ void stage_f32_to_f16(const float* __restrict__ g,
                                                 int gstride, _Float16* lds,
                                                 int rows, int t) {
  const int chunks = rows * 8;  // 4 elements per chunk
  for (int c = t; c < chunks; c += 256) {
    int row = c >> 3, c4 = (c & 7) * 4;
    v4f xv = *(const v4f*)(g + (size_t)row * gstride + c4);
    v4h hv;
    hv[0] = (_Float16)xv[0]; hv[1] = (_Float16)xv[1];
    hv[2] = (_Float16)xv[2]; hv[3] = (_Float16)xv[3];
    *(v4h*)(lds + row * 32 + c4) = hv;
  }
}

// ---------------------------------------------------------------------------
// Kernel A: Y[m,n] = sum_k X[m,k]*W[n,k] + bias[n]; emit f16 heads [B,H,S,DK].
// 128x64 tile per workgroup, 8 waves, 4 accumulators each, K-step 32.
// ---------------------------------------------------------------------------
__global__ __launch_bounds__(256) void proj_kernel(
    const float* __restrict__ x, const float* __restrict__ W,
    const float* __restrict__ bias, _Float16* __restrict__ out_h) {
  __shared__ _Float16 ldsA[128 * 32];
  __shared__ _Float16 ldsW[64 * 32];
  const int t = threadIdx.x, lane = t & 31, wave = t >> 5;
  const int m0 = blockIdx.x * 128, n0 = blockIdx.y * 64;

  v8f acc[4] = {};
  for (int k0 = 0; k0 < D_DIM; k0 += 32) {
    __syncthreads();
    stage_f32_to_f16(x + (size_t)m0 * D_DIM + k0, D_DIM, ldsA, 128, t);
    stage_f32_to_f16(W + (size_t)n0 * D_DIM + k0, D_DIM, ldsW, 64, t);
    __syncthreads();
    v16h a = load_a_frag(ldsA + wave * 16 * 32, 32, lane);
#pragma unroll
    for (int nt = 0; nt < 4; ++nt) {
      v16h b = load_b_frag(ldsW + nt * 16 * 32, 32, lane);
      acc[nt] = wmma16(a, b, acc[nt]);
    }
  }
  // C/D layout: VGPR r -> M = r + 8*(lane>>4), N = lane&15
  const int mhi = (lane >> 4) * 8, nc = lane & 15;
#pragma unroll
  for (int nt = 0; nt < 4; ++nt) {
    int n = n0 + nt * 16 + nc;
    int h = n >> 6, dk = n & 63;
    float bn = bias[n];
#pragma unroll
    for (int r = 0; r < 8; ++r) {
      int m = m0 + wave * 16 + mhi + r;
      int bi = m >> 10, s = m & 1023;
      out_h[(size_t)((bi * H_DIM + h) * S_DIM + s) * DK_DIM + dk] =
          (_Float16)(acc[nt][r] + bn);
    }
  }
}

// ---------------------------------------------------------------------------
// Kernel B: one workgroup per (b,h, 32-query block).
// scores(LDS f32) -> fused scale/bias/mask -> softmax -> attn f32 to HBM
// (+f16 LDS) -> attn @ V with async-DMA-staged V tiles -> ctx f16 [B,S,D].
// Dynamic LDS: 128K scores + 64K attn(f16) + 9K vT + 8K vstage + 4K bias tab.
// ---------------------------------------------------------------------------
#define VT_STRIDE 72  // padded row stride (halves): 4-way instead of 32-way conflicts
#define SMEM_ATTN (131072 + 65536 + 9216 + 8192 + 4096)  // 218,112B (<320KB/WGP)

__global__ __launch_bounds__(256) void attn_kernel(
    const _Float16* __restrict__ qh, const _Float16* __restrict__ kh,
    const _Float16* __restrict__ vh, const float* __restrict__ bias_p,
    const float* __restrict__ bias_a, const int* __restrict__ zero_pad,
    float* __restrict__ attn_out, _Float16* __restrict__ ctx) {
  extern __shared__ char smem[];
  float*    scores = (float*)smem;                        // 32*1024 f32
  _Float16* attnh  = (_Float16*)(smem + 131072);          // 32*1024 f16
  _Float16* vT     = (_Float16*)(smem + 196608);          // 64*VT_STRIDE f16
  _Float16* vstage = (_Float16*)(smem + 205824);          // 64*64 f16 (async dest)
  float*    btab   = (float*)(smem + 214016);             // 1024 f32

  const int t = threadIdx.x, lane = t & 31, wave = t >> 5;
  const int qblk = blockIdx.x & 31;
  const int bh   = blockIdx.x >> 5;
  const int h    = bh & (H_DIM - 1);
  const int q0   = qblk * 32;
  const _Float16* qB = qh + (size_t)bh * S_DIM * DK_DIM;
  const _Float16* kB = kh + (size_t)bh * S_DIM * DK_DIM;
  const _Float16* vB = vh + (size_t)bh * S_DIM * DK_DIM;

  // gfx1250 async global->LDS DMA of one V tile (64 keys x 64 dk, contiguous
  // 8KB in global). 512 x 16B chunks over 256 lanes; tracked by ASYNCcnt.
  const unsigned vstage_base = lds_off(vstage);
  auto issue_vstage = [&](int kt0) {
    const _Float16* src = vB + (size_t)kt0 * DK_DIM;
#pragma unroll
    for (int iu = 0; iu < 2; ++iu) {
      int ch = t + 256 * iu;
      unsigned loff = vstage_base + ch * 16;
      unsigned long long ga = (unsigned long long)(uintptr_t)(src + ch * 8);
      asm volatile("global_load_async_to_lds_b128 %0, %1, off"
                   :: "v"(loff), "v"(ga) : "memory");
    }
  };
  issue_vstage(0);  // overlap first V tile DMA with the whole score phase

  // Kerple log-distance bias table: -p*log1p(a*d), d=0..1023
  {
    float p = fmaxf(bias_p[h], 0.01f);
    float a = fmaxf(bias_a[h], 0.01f);
    for (int d = t; d < S_DIM; d += 256) btab[d] = -p * log1pf(a * (float)d);
  }
  __syncthreads();

  // ---- scores: tiles only up to the causal diagonal -----------------------
  const int ntcnt = 2 * qblk + 2;  // key tiles needed for rows q0..q0+31
  for (int tid = wave; tid < 2 * ntcnt; tid += 8) {
    int mt = tid & 1, nt = tid >> 1;
    const _Float16* qb = qB + (size_t)(q0 + mt * 16) * DK_DIM;
    const _Float16* kb = kB + (size_t)(nt * 16) * DK_DIM;
    v8f c = {};
#pragma unroll
    for (int ks = 0; ks < DK_DIM; ks += 32) {
      v16h a = load_a_frag(qb + ks, DK_DIM, lane);  // q rows (contiguous dk)
      v16h b = load_b_frag(kb + ks, DK_DIM, lane);  // k rows == B[k=dk][n=key]
      c = wmma16(a, b, c);
    }
    int mhi = (lane >> 4) * 8, nc = lane & 15;
#pragma unroll
    for (int r = 0; r < 8; ++r) {
      int li = mt * 16 + mhi + r;
      int i  = q0 + li;                 // global query index
      int j  = nt * 16 + nc;            // global key index
      scores[li * S_DIM + j] =
          (j > i) ? -1.0e32f : (c[r] * 0.125f + btab[i - j]);  // 1/sqrt(64)
    }
  }
  __syncthreads();

  // ---- softmax per row (wave32 shuffle reduce), write attn ---------------
  const bool dz = (zero_pad[0] != 0);
#pragma unroll
  for (int rr = 0; rr < 4; ++rr) {
    int li = wave * 4 + rr;
    int i  = q0 + li;
    float* srow = scores + li * S_DIM;
    float mx = -3.0e38f;
    for (int j = lane; j <= i; j += 32) mx = fmaxf(mx, srow[j]);
#pragma unroll
    for (int off = 16; off > 0; off >>= 1) mx = fmaxf(mx, __shfl_xor(mx, off, 32));
    float sum = 0.0f;
    for (int j = lane; j <= i; j += 32) {
      float e = __expf(srow[j] - mx);
      srow[j] = e;
      sum += e;
    }
#pragma unroll
    for (int off = 16; off > 0; off >>= 1) sum += __shfl_xor(sum, off, 32);
    float inv = 1.0f / sum;
    bool zrow = dz && (i == 0);  // reference zeroes the first query row
    float*    arow = attn_out + ((size_t)(bh * S_DIM + i) << 10);
    _Float16* hrow = attnh + li * S_DIM;
    for (int jb = lane * 4; jb < S_DIM; jb += 128) {  // b128 stores of attn
      v4f av; v4h hv;
#pragma unroll
      for (int c4 = 0; c4 < 4; ++c4) {
        int j = jb + c4;
        float av1 = (!zrow && j <= i) ? srow[j] * inv : 0.0f;
        av[c4] = av1; hv[c4] = (_Float16)av1;
      }
      *(v4f*)(arow + jb) = av;   // the single mandatory HBM write of attn
      *(v4h*)(hrow + jb) = hv;
    }
  }

  // ---- out = attn @ V; keys beyond q0+31 contribute exactly 0 ------------
  v8f oacc = {};
  const int mt = wave & 1, nt = wave >> 1;  // 2 M-tiles x 4 dk-tiles = 8 waves
  const int klim = q0 + 32;
  for (int kt0 = 0; kt0 < klim; kt0 += 64) {
    asm volatile("s_wait_asynccnt 0x0" ::: "memory");  // this wave's DMA done
    __syncthreads();                                   // all waves' DMA visible
    // transpose vstage [key][dk] -> vT [dk][key] (coalesced reads, padded rows)
#pragma unroll
    for (int iu = 0; iu < 16; ++iu) {
      int e = t + 256 * iu;
      int kk = e >> 6, dk = e & 63;
      vT[dk * VT_STRIDE + kk] = vstage[kk * 64 + dk];
    }
    __syncthreads();                                   // vT ready, vstage free
    if (kt0 + 64 < klim) issue_vstage(kt0 + 64);       // pipeline next tile DMA
#pragma unroll
    for (int ks = 0; ks < 64; ks += 32) {
      v16h a = load_a_frag(attnh + mt * 16 * S_DIM + kt0 + ks, S_DIM, lane);
      v16h b = load_b_frag_u(vT + nt * 16 * VT_STRIDE + ks, VT_STRIDE, lane);
      oacc = wmma16(a, b, oacc);
    }
  }
  const int mhi = (lane >> 4) * 8, nc = lane & 15;
  const int bi = bh >> 3;
#pragma unroll
  for (int r = 0; r < 8; ++r) {
    int qg = q0 + mt * 16 + mhi + r;
    int f  = h * DK_DIM + nt * 16 + nc;      // head-concat feature index
    ctx[(size_t)(bi * S_DIM + qg) * D_DIM + f] = (_Float16)oacc[r];
  }
}

// ---------------------------------------------------------------------------
// Kernel C: output = ctx(f16) @ Wo^T + bo, f32 result.
// ---------------------------------------------------------------------------
__global__ __launch_bounds__(256) void outproj_kernel(
    const _Float16* __restrict__ ctx, const float* __restrict__ Wo,
    const float* __restrict__ bo, float* __restrict__ out) {
  __shared__ _Float16 ldsW[64 * 32];
  const int t = threadIdx.x, lane = t & 31, wave = t >> 5;
  const int m0 = blockIdx.x * 128, n0 = blockIdx.y * 64;
  v8f acc[4] = {};
  for (int k0 = 0; k0 < D_DIM; k0 += 32) {
    __syncthreads();
    stage_f32_to_f16(Wo + (size_t)n0 * D_DIM + k0, D_DIM, ldsW, 64, t);
    __syncthreads();
    v16h a = load_a_frag(ctx + (size_t)(m0 + wave * 16) * D_DIM + k0, D_DIM, lane);
#pragma unroll
    for (int nt = 0; nt < 4; ++nt) {
      v16h b = load_b_frag(ldsW + nt * 16 * 32, 32, lane);
      acc[nt] = wmma16(a, b, acc[nt]);
    }
  }
  const int mhi = (lane >> 4) * 8, nc = lane & 15;
#pragma unroll
  for (int nt = 0; nt < 4; ++nt) {
    int n = n0 + nt * 16 + nc;
    float bn = bo[n];
#pragma unroll
    for (int r = 0; r < 8; ++r) {
      int m = m0 + wave * 16 + mhi + r;
      out[(size_t)m * D_DIM + n] = acc[nt][r] + bn;
    }
  }
}

// ---------------------------------------------------------------------------
extern "C" void kernel_launch(void* const* d_in, const int* in_sizes, int n_in,
                              void* d_out, int out_size, void* d_ws, size_t ws_size,
                              hipStream_t stream) {
  (void)in_sizes; (void)n_in; (void)out_size; (void)ws_size;

  const float* q  = (const float*)d_in[0];
  const float* k  = (const float*)d_in[1];
  const float* v  = (const float*)d_in[2];
  /* d_in[3] = causal mask, implied by the kernel's j<=i test */
  const float* Wq = (const float*)d_in[4];
  const float* bq = (const float*)d_in[5];
  const float* Wk = (const float*)d_in[6];
  const float* bk = (const float*)d_in[7];
  const float* Wv = (const float*)d_in[8];
  const float* bv = (const float*)d_in[9];
  const float* Wo = (const float*)d_in[10];
  const float* bo = (const float*)d_in[11];
  const float* bp = (const float*)d_in[12];
  const float* ba = (const float*)d_in[13];
  const int*   zp = (const int*)d_in[14];

  float* out  = (float*)d_out;
  float* attn = out + (size_t)B_DIM * S_DIM * D_DIM;  // outputs concat: [out, attn]

  // Workspace: 3x f16 head tensors (8MB each) + f16 context (8MB) = 32MB.
  const size_t HE = (size_t)B_DIM * H_DIM * S_DIM * DK_DIM;
  _Float16* qh  = (_Float16*)d_ws;
  _Float16* kh  = qh + HE;
  _Float16* vh  = kh + HE;
  _Float16* ctx = vh + HE;

  dim3 gp(64, 8);  // 8192/128 x 512/64
  proj_kernel<<<gp, 256, 0, stream>>>(q, Wq, bq, qh);
  proj_kernel<<<gp, 256, 0, stream>>>(k, Wk, bk, kh);
  proj_kernel<<<gp, 256, 0, stream>>>(v, Wv, bv, vh);

  hipFuncSetAttribute(reinterpret_cast<const void*>(attn_kernel),
                      hipFuncAttributeMaxDynamicSharedMemorySize, SMEM_ATTN);
  attn_kernel<<<dim3(B_DIM * H_DIM * (S_DIM / 32)), 256, SMEM_ATTN, stream>>>(
      qh, kh, vh, bp, ba, zp, attn, ctx);

  outproj_kernel<<<gp, 256, 0, stream>>>(ctx, Wo, bo, out);
}
